// StepwiseAttention_28389733827180
// MI455X (gfx1250) — matrix operand order repaired and checked by reading, not compiled, not run on hardware
//
#include <hip/hip_runtime.h>
#include <hip/hip_bf16.h>
#include <math.h>

// Problem dims (from reference)
#define BATCH 64
#define SEQ   4096
#define HID   512
#define ATT   256
#define INP   256

// Stage-B decomposition: 64 batches x 32 splits = 2048 blocks x 2 waves
// = 4096 waves (~4 per SIMD32) for latency hiding.
#define NSPLIT 32
#define SCHUNK (SEQ / NSPLIT)          // 128 seq positions per block
#define TILES_PER_BLOCK (SCHUNK / 16)  // 8 tiles of 16 rows
#define WAVES 2
#define TB_THREADS (WAVES * 32)        // 64 threads, wave32

typedef __bf16 v16bf __attribute__((ext_vector_type(16)));
typedef __bf16 v8bf  __attribute__((ext_vector_type(8)));
typedef __bf16 v4bf  __attribute__((ext_vector_type(4)));
typedef float  v8f   __attribute__((ext_vector_type(8)));
typedef float  f32x4 __attribute__((ext_vector_type(4)));
typedef unsigned int u32;
typedef u32 u32x4 __attribute__((ext_vector_type(4)));

#if defined(__AMDGCN__) && __has_builtin(__builtin_amdgcn_sched_barrier)
#define SCHED_FENCE() __builtin_amdgcn_sched_barrier(0)
#else
#define SCHED_FENCE()
#endif

// ---------------------------------------------------------------------------
// Stage A: Q = x_t@Wq + bq ; qk[b,h] = (Wk@Q)/16 (bf16) ; qbk[b] = Q.bk/16
// Tiny (~25 MFLOP) — plain VALU is fine.
// ---------------------------------------------------------------------------
__global__ __launch_bounds__(256)
void qk_precompute(const float* __restrict__ x_t,
                   const float* __restrict__ Wq,
                   const float* __restrict__ bq,
                   const float* __restrict__ Wk,
                   const float* __restrict__ bk,
                   __bf16* __restrict__ qk_out,
                   float*  __restrict__ qbk_out) {
  const int b = blockIdx.x;
  const int a = threadIdx.x;           // 0..255
  __shared__ float Qs[ATT];
  __shared__ float red[256];

  float q = bq[a];
  const float* xr = x_t + (size_t)b * INP;
  for (int i = 0; i < INP; ++i) q = fmaf(xr[i], Wq[i * ATT + a], q);
  Qs[a]  = q;
  red[a] = q * bk[a];
  __syncthreads();

  for (int off = 128; off > 0; off >>= 1) {
    if (a < off) red[a] += red[a + off];
    __syncthreads();
  }
  if (a == 0) qbk_out[b] = red[0] * 0.0625f;   // 1/sqrt(256)

  for (int h = a; h < HID; h += 256) {
    float s = 0.f;
    const float* wr = Wk + (size_t)h * ATT;
    for (int i = 0; i < ATT; ++i) s = fmaf(Qs[i], wr[i], s);
    qk_out[(size_t)b * HID + h] = (__bf16)(s * 0.0625f);
  }
}

// ---------------------------------------------------------------------------
// Stage B: single streaming pass over history. Per 16-row s-tile:
//   scores = WMMA( H_tile(16x512 bf16), qk broadcast into all 16 cols )
//   online softmax (m, l), weighted accumulation acc[h] += p_s * H[s][h]
// Partials {m, l, acc[512]} per (batch, split) written to workspace.
// ---------------------------------------------------------------------------
__global__ __launch_bounds__(TB_THREADS)
void attn_stream(const float*  __restrict__ hist,
                 const __bf16* __restrict__ qk,
                 const float*  __restrict__ qbk,
                 float*        __restrict__ partials) {
  const int split = blockIdx.x;
  const int b     = blockIdx.y;
  const int tid   = threadIdx.x;
  const int wave  = tid >> 5;
  const int lane  = tid & 31;

  // 520-element row stride (16B pad) keeps WMMA A-fragment reads conflict-free
  __shared__ __align__(64) __bf16 tileA[WAVES][16][520];  // 33,280 B
  __shared__ __align__(64) __bf16 qks[HID];               //  1,024 B
  __shared__ float accsh[WAVES][HID];                     //  4,096 B
  __shared__ float mw[WAVES], lw[WAVES];

  // Stage qk into LDS (32-bit copies)
  {
    const u32* s = (const u32*)(qk + (size_t)b * HID);
    u32* d = (u32*)qks;
    for (int i = tid; i < HID / 2; i += TB_THREADS) d[i] = s[i];
  }
  const float qbk_b = qbk[b];
  __syncthreads();

  const int kbase = (lane >> 4) * 16;  // A-fragment K-base per half-wave
  const int mrow  = lane & 15;         // A-fragment row
  const int boff  = (lane >> 4) * 16;  // B-fragment K-base per half-wave

  // Preload ALL 16 B fragments (qk broadcast) into registers — they are
  // invariant across every tile this wave processes.  128 VGPRs, by design.
  v16bf bfr[16];
  #pragma unroll
  for (int ks = 0; ks < 16; ++ks)
    bfr[ks] = *(const v16bf*)&qks[ks * 32 + boff];

  float m = -INFINITY, l = 0.f;
  float acc[16];                       // h = lane*16 + j  (covers all 512)
  #pragma unroll
  for (int j = 0; j < 16; ++j) acc[j] = 0.f;

  for (int t = wave; t < TILES_PER_BLOCK; t += WAVES) {
    const size_t srow = (size_t)split * SCHUNK + (size_t)t * 16;
    const float* src  = hist + ((size_t)b * SEQ + srow) * HID;

    // ---- Phase 1: stage 16x512 f32 tile -> bf16 LDS (coalesced, NT) ----
    #pragma unroll 8
    for (int it = 0; it < 64; ++it) {
      const int flat = it * 32 + lane;          // 2048 float4 slots
      const int row  = flat >> 7;               // 128 float4 per row
      const int c4   = flat & 127;
      f32x4 v = __builtin_nontemporal_load((const f32x4*)(src + row * HID + c4 * 4));
      v4bf bv4 = { (__bf16)v[0], (__bf16)v[1], (__bf16)v[2], (__bf16)v[3] };
      *(v4bf*)&tileA[wave][row][c4 * 4] = bv4;
    }
    // Same-wave DS ops are in-order; no barrier needed (per-wave LDS region).

    // ---- Prefetch this wave's NEXT tile into L2 while we compute ----
    // (32KB = 256 cachelines; 8 instrs x 32 lanes x 128B)
    if (t + WAVES < TILES_PER_BLOCK) {
      const float* nsrc = src + (size_t)WAVES * 16 * HID;
      #pragma unroll
      for (int pf = 0; pf < 8; ++pf)
        __builtin_prefetch(nsrc + (size_t)pf * 1024 + lane * 32, 0, 3);
    }

    // ---- Phase 2: scores via v_wmma_f32_16x16x32_bf16, K = 512 ----
    // A loads pipelined one 4-fragment group ahead; B comes from registers,
    // so each WMMA group runs back-to-back with only the A-group wait.
    auto loadA = [&](int ks) -> v16bf {
      const int kb = ks * 32;
      v8bf alo = *(const v8bf*)&tileA[wave][mrow][kb + kbase];
      v8bf ahi = *(const v8bf*)&tileA[wave][mrow][kb + kbase + 8];
      return __builtin_shufflevector(alo, ahi,
          0, 1, 2, 3, 4, 5, 6, 7, 8, 9, 10, 11, 12, 13, 14, 15);
    };
    v8f c = {0.f, 0.f, 0.f, 0.f, 0.f, 0.f, 0.f, 0.f};
    v16bf f0 = loadA(0), f1 = loadA(1), f2 = loadA(2), f3 = loadA(3);
    #pragma unroll
    for (int g = 0; g < 4; ++g) {
      v16bf n0, n1, n2, n3;
      if (g < 3) {
        n0 = loadA(4 * g + 4);
        n1 = loadA(4 * g + 5);
        n2 = loadA(4 * g + 6);
        n3 = loadA(4 * g + 7);
      }
      SCHED_FENCE();   // keep next-group loads issued above this group's WMMAs
      c = __builtin_amdgcn_wmma_f32_16x16x32_bf16(false, f0, false, bfr[4 * g + 0], (short)0, c, false, false);
      c = __builtin_amdgcn_wmma_f32_16x16x32_bf16(false, f1, false, bfr[4 * g + 1], (short)0, c, false, false);
      c = __builtin_amdgcn_wmma_f32_16x16x32_bf16(false, f2, false, bfr[4 * g + 2], (short)0, c, false, false);
      c = __builtin_amdgcn_wmma_f32_16x16x32_bf16(false, f3, false, bfr[4 * g + 3], (short)0, c, false, false);
      SCHED_FENCE();   // pin the load/WMMA interleave pattern
      f0 = n0; f1 = n1; f2 = n2; f3 = n3;
    }

    // ---- Online softmax update (scores = c[r] + qbk_b, scale pre-folded) ----
    float tmax = c[0];
    #pragma unroll
    for (int r = 1; r < 8; ++r) tmax = fmaxf(tmax, c[r]);
    tmax = fmaxf(tmax, __shfl_xor(tmax, 16, 32));     // rows 0-7 vs 8-15
    const float mnew  = fmaxf(m, tmax + qbk_b);
    const float alpha = __expf(m - mnew);
    float p[8], rs = 0.f;
    #pragma unroll
    for (int r = 0; r < 8; ++r) { p[r] = __expf(c[r] + qbk_b - mnew); rs += p[r]; }
    rs += __shfl_xor(rs, 16, 32);
    l = l * alpha + rs;
    m = mnew;
    #pragma unroll
    for (int j = 0; j < 16; ++j) acc[j] *= alpha;

    // Assemble all 16 row-probabilities in every lane
    float pall[16];
    const bool lowhalf = (lane < 16);
    #pragma unroll
    for (int r = 0; r < 8; ++r) {
      float o = __shfl_xor(p[r], 16, 32);
      pall[r]     = lowhalf ? p[r] : o;
      pall[r + 8] = lowhalf ? o : p[r];
    }

    // ---- Phase 3: weighted accumulation of the raw history rows ----
    // h = lane*16 + j ; two ds_load_b128 per row per lane.
    #pragma unroll 4
    for (int s = 0; s < 16; ++s) {
      const u32x4* rowp =
          (const u32x4*)((const u32*)&tileA[wave][s][0] + lane * 8);
      const u32x4 w0 = rowp[0];
      const u32x4 w1 = rowp[1];
      const float ps = pall[s];
      #pragma unroll
      for (int q = 0; q < 4; ++q) {
        const u32 wa = w0[q];
        acc[2 * q]     = fmaf(ps, __uint_as_float(wa << 16),          acc[2 * q]);
        acc[2 * q + 1] = fmaf(ps, __uint_as_float(wa & 0xFFFF0000u), acc[2 * q + 1]);
        const u32 wb = w1[q];
        acc[8 + 2 * q]     = fmaf(ps, __uint_as_float(wb << 16),          acc[8 + 2 * q]);
        acc[8 + 2 * q + 1] = fmaf(ps, __uint_as_float(wb & 0xFFFF0000u), acc[8 + 2 * q + 1]);
      }
    }
  }

  // ---- Combine the 2 waves (log-sum-exp merge) and write partial ----
  if (lane == 0) { mw[wave] = m; lw[wave] = l; }
  #pragma unroll
  for (int j = 0; j < 16; ++j) accsh[wave][lane * 16 + j] = acc[j];
  __syncthreads();

  const float M  = fmaxf(mw[0], mw[1]);
  const float e0 = __expf(mw[0] - M);
  const float e1 = __expf(mw[1] - M);
  float* P = partials + ((size_t)b * NSPLIT + split) * (HID + 2);
  if (tid == 0) { P[0] = M; P[1] = lw[0] * e0 + lw[1] * e1; }
  for (int h = tid; h < HID; h += TB_THREADS)
    P[2 + h] = accsh[0][h] * e0 + accsh[1][h] * e1;
}

// ---------------------------------------------------------------------------
// Stage C: merge NSPLIT partials per batch, then context = w @ Wv + bv
// ---------------------------------------------------------------------------
__global__ __launch_bounds__(256)
void finalize(const float* __restrict__ partials,
              const float* __restrict__ Wv,
              const float* __restrict__ bv,
              float*       __restrict__ out) {
  const int b = blockIdx.x;
  const int a = threadIdx.x;           // 0..255
  __shared__ float pm[NSPLIT], pl[NSPLIT];
  __shared__ float wsum[HID];
  __shared__ float esh[NSPLIT];
  const float* Pb = partials + (size_t)b * NSPLIT * (HID + 2);

  if (a < NSPLIT) {
    pm[a] = Pb[(size_t)a * (HID + 2)];
    pl[a] = Pb[(size_t)a * (HID + 2) + 1];
  }
  __syncthreads();

  float M = -INFINITY;
  #pragma unroll
  for (int i = 0; i < NSPLIT; ++i) M = fmaxf(M, pm[i]);
  if (a < NSPLIT) esh[a] = __expf(pm[a] - M);
  __syncthreads();
  float L = 0.f;
  #pragma unroll
  for (int i = 0; i < NSPLIT; ++i) L = fmaf(pl[i], esh[i], L);
  const float invL = 1.0f / L;

  for (int h = a; h < HID; h += 256) {
    float s = 0.f;
    #pragma unroll
    for (int i = 0; i < NSPLIT; ++i)
      s = fmaf(Pb[(size_t)i * (HID + 2) + 2 + h], esh[i], s);
    wsum[h] = s * invL;
  }
  __syncthreads();

  float ctx = bv[a];
  for (int h = 0; h < HID; ++h) ctx = fmaf(wsum[h], Wv[h * ATT + a], ctx);
  out[(size_t)b * ATT + a] = ctx;
}

// ---------------------------------------------------------------------------
extern "C" void kernel_launch(void* const* d_in, const int* in_sizes, int n_in,
                              void* d_out, int out_size, void* d_ws, size_t ws_size,
                              hipStream_t stream) {
  const float* x_t  = (const float*)d_in[0];
  const float* hist = (const float*)d_in[1];
  const float* Wq   = (const float*)d_in[2];
  const float* bq   = (const float*)d_in[3];
  const float* Wk   = (const float*)d_in[4];
  const float* bk   = (const float*)d_in[5];
  const float* Wv   = (const float*)d_in[6];
  const float* bv   = (const float*)d_in[7];

  char* ws = (char*)d_ws;
  __bf16* qk    = (__bf16*)ws;             // 64*512*2  = 65,536 B
  float*  qbk   = (float*)(ws + 65536);    // 64*4      =    256 B
  float*  parts = (float*)(ws + 65792);    // 64*32*514*4 ≈ 4.2 MB

  qk_precompute<<<BATCH, 256, 0, stream>>>(x_t, Wq, bq, Wk, bk, qk, qbk);
  attn_stream<<<dim3(NSPLIT, BATCH), TB_THREADS, 0, stream>>>(hist, qk, qbk, parts);
  finalize<<<BATCH, 256, 0, stream>>>(parts, Wv, bv, (float*)d_out);
}
